// SignNet_86363202388258
// MI455X (gfx1250) — compile-verified, matrix operand order checked by reading, not run on hardware
//
#include <hip/hip_runtime.h>

// ---------------------------------------------------------------------------
// CDNA5 (gfx1250) fused GINE / SignNet implementation.
// Wave32, WMMA f16->f32 16x16x32 for all matmuls.
// Weights pre-packed once into per-lane WMMA fragment layout (f16); hot
// kernels fetch whole B fragments with 32B vector loads. Edge scatter uses
// hoisted row pointers + immediate offsets, batched gathers, scalar bounds.
// ---------------------------------------------------------------------------

typedef __attribute__((ext_vector_type(16))) _Float16 v16h;
typedef __attribute__((ext_vector_type(8)))  float    v8f;

// K index held by (vgpr v, half p) of this lane for a 16-bit 16x32 A (or 32x16
// B) fragment, per CDNA5 ISA 7.12.2.
__device__ __forceinline__ int wmma_k_idx(int lane, int v, int p) {
    return ((v & 3) * 2 + p) + ((v >> 2) * 16) + ((lane >> 4) * 8);
}

// ---------------------------------------------------------------------------
// Weight pre-pack: W [K x NC] (f32, row-major) -> f16 fragments laid out as
//   dst[((kt*NT + t)*32 + lane)*16 + j]
// Optionally writes bias[n] into virtual row `bias_row` (zero-padded K space).
// ---------------------------------------------------------------------------
__global__ void pack_frag_kernel(const float* __restrict__ W,
                                 const float* __restrict__ bias,
                                 _Float16* __restrict__ dst,
                                 int K, int NC, int KT, int NT, int bias_row)
{
    const int idx = blockIdx.x * blockDim.x + threadIdx.x;
    const int total = KT * NT * 32 * 16;
    if (idx >= total) return;
    const int j    = idx & 15;
    const int lane = (idx >> 4) & 31;
    const int frag = idx >> 9;           // kt*NT + t
    const int t    = frag % NT;
    const int kt   = frag / NT;
    const int v = j >> 1, p = j & 1;
    const int k = kt * 32 + wmma_k_idx(lane, v, p);
    const int n = t * 16 + (lane & 15);
    float val = 0.0f;
    if (k < K)                       val = W[(size_t)k * NC + n];
    else if (bias && k == bias_row)  val = bias[n];
    dst[idx] = (_Float16)val;
}

// ---------------------------------------------------------------------------
// Fused edge kernel (FULL tiles only): e = edge_attr @ we + be (bias folded
// into packed we at K=16 via a constant-1.0 A element), then
//   msg = relu(h[src] + e);  agg[dst] += msg   (L2 atomics)
// ---------------------------------------------------------------------------
template<int DIN>
__global__ __launch_bounds__(256)
void edge_msg_kernel(const float* __restrict__ h,
                     const float* __restrict__ edge_attr,
                     const int*   __restrict__ esrc,
                     const int*   __restrict__ edst,
                     const _Float16* __restrict__ we_pack,  // NT frags, bias@K=16
                     float*       __restrict__ agg,
                     int Efull)                             // multiple of 16
{
    constexpr int NT = DIN / 16;
    const int lane = threadIdx.x & 31;
    // wave tile start, forced into an SGPR -> scalar bounds branch
    const int e0 = __builtin_amdgcn_readfirstlane(
        ((blockIdx.x * blockDim.x + threadIdx.x) >> 5) * 16);
    if (e0 >= Efull) return;

    // ---- A fragment: 16 edges x K16 (padded to 32; K=16 element = 1.0) ----
    const int arow = lane & 15;
    const int off = (lane >> 4) * 8;
    const float4* ea4 = (const float4*)(edge_attr + (size_t)(e0 + arow) * 16 + off);
    const float4 f0 = ea4[0];
    const float4 f1 = ea4[1];
    v16h a;
    a[0] = (_Float16)f0.x; a[1] = (_Float16)f0.y;
    a[2] = (_Float16)f0.z; a[3] = (_Float16)f0.w;
    a[4] = (_Float16)f1.x; a[5] = (_Float16)f1.y;
    a[6] = (_Float16)f1.z; a[7] = (_Float16)f1.w;
    const _Float16 one = (_Float16)1.0f, zer = (_Float16)0.0f;
    a[8] = (lane < 16) ? one : zer;        // K=16 -> bias row of we_pack
#pragma unroll
    for (int j = 9; j < 16; ++j) a[j] = zer;

    // ---- hoisted row base pointers for this lane's 8 C elements ----
    const int mbase = (lane >> 4) * 8;
    const int col = lane & 15;
    const float* hp[8];
    float*       ap[8];
#pragma unroll
    for (int i = 0; i < 8; ++i) {
        const int ei = e0 + mbase + i;
        hp[i] = h   + (size_t)esrc[ei] * DIN + col;
        ap[i] = agg + (size_t)edst[ei] * DIN + col;
    }

    // ---- WMMA + batched gather + relu + scatter per 16-wide column tile ----
#pragma unroll
    for (int t = 0; t < NT; ++t) {
        const v16h b = *(const v16h*)(we_pack + ((size_t)t * 32 + lane) * 16);
        v8f c = {};
        c = __builtin_amdgcn_wmma_f32_16x16x32_f16(
                false, a, false, b, (short)0, c, false, false);
        float hv[8];
#pragma unroll
        for (int i = 0; i < 8; ++i)            // batch the 8 gathers (clause)
            hv[i] = hp[i][t * 16];
#pragma unroll
        for (int i = 0; i < 8; ++i) {
            float msg = hv[i] + c[i];
            msg = msg > 0.0f ? msg : 0.0f;
            atomicAdd(ap[i] + t * 16, msg);    // imm-offset atomic
        }
    }
}

// ---------------------------------------------------------------------------
// Tail edges (E % 16 != 0 only; never launched for this workload's shapes).
// One thread per (edge, feature): scalar f32 path.
// ---------------------------------------------------------------------------
__global__ void edge_tail_kernel(const float* __restrict__ h,
                                 const float* __restrict__ edge_attr,
                                 const int*   __restrict__ esrc,
                                 const int*   __restrict__ edst,
                                 const float* __restrict__ we,
                                 const float* __restrict__ be,
                                 float*       __restrict__ agg,
                                 int DIN, int e_start, int E)
{
    const int idx = blockIdx.x * blockDim.x + threadIdx.x;
    const int total = (E - e_start) * DIN;
    if (idx >= total) return;
    const int e = e_start + idx / DIN;
    const int n = idx % DIN;
    float acc = be[n];
    for (int k = 0; k < 16; ++k)
        acc += edge_attr[(size_t)e * 16 + k] * we[k * DIN + n];
    float msg = h[(size_t)esrc[e] * DIN + n] + acc;
    msg = msg > 0.0f ? msg : 0.0f;
    atomicAdd(&agg[(size_t)edst[e] * DIN + n], msg);
}

// ---------------------------------------------------------------------------
// Node MLP kernel:  out = [relu]( relu((hin+agg) @ w1 + b1) @ w2 + b2 )
// One wave per 16 nodes; hidden bounces through LDS for C->A re-layout.
// ---------------------------------------------------------------------------
template<int DIN, int DH, int DOUT, bool RELU_OUT>
__global__ __launch_bounds__(128)
void node_mlp_kernel(const float* __restrict__ hin,
                     const float* __restrict__ agg,
                     const _Float16* __restrict__ w1p, const float* __restrict__ b1,
                     const _Float16* __restrict__ w2p, const float* __restrict__ b2,
                     float*       __restrict__ hout,
                     int Nn)
{
    constexpr int WV  = 4;
    constexpr int KT1 = (DIN + 31) / 32;
    constexpr int NT1 = DH / 16;
    constexpr int KT2 = DH / 32;
    constexpr int NT2 = DOUT / 16;

    __shared__ float lds_h[WV][16][DH];

    const int wave = threadIdx.x >> 5;
    const int lane = threadIdx.x & 31;
    // scalar row-tile start (clamped; duplicate work instead of early return)
    int r0 = __builtin_amdgcn_readfirstlane((blockIdx.x * WV + wave) * 16);
    if (r0 + 16 > Nn) r0 = (Nn >= 16) ? (Nn - 16) : 0;

    const int arow  = lane & 15;
    const int col   = lane & 15;
    const int mbase = (lane >> 4) * 8;
    const int off   = (lane >> 4) * 8;

    const float* hrow = hin + (size_t)(r0 + arow) * DIN;
    const float* grow = agg + (size_t)(r0 + arow) * DIN;

    // ---------------- matmul 1: (hin+agg) @ w1 ----------------
    v8f acc1[NT1];
#pragma unroll
    for (int t = 0; t < NT1; ++t) acc1[t] = (v8f){};

#pragma unroll
    for (int kt = 0; kt < KT1; ++kt) {
        v16h aF;
        const int kA = kt * 32 + off;
        const int kB = kt * 32 + 16 + off;
        if (kA < DIN) {
            const float4* hp4 = (const float4*)(hrow + kA);
            const float4* gp4 = (const float4*)(grow + kA);
            float4 h0 = hp4[0], h1 = hp4[1], g0 = gp4[0], g1 = gp4[1];
            aF[0] = (_Float16)(h0.x + g0.x); aF[1] = (_Float16)(h0.y + g0.y);
            aF[2] = (_Float16)(h0.z + g0.z); aF[3] = (_Float16)(h0.w + g0.w);
            aF[4] = (_Float16)(h1.x + g1.x); aF[5] = (_Float16)(h1.y + g1.y);
            aF[6] = (_Float16)(h1.z + g1.z); aF[7] = (_Float16)(h1.w + g1.w);
        } else {
#pragma unroll
            for (int j = 0; j < 8; ++j) aF[j] = (_Float16)0.0f;
        }
        if (kB < DIN) {
            const float4* hp4 = (const float4*)(hrow + kB);
            const float4* gp4 = (const float4*)(grow + kB);
            float4 h0 = hp4[0], h1 = hp4[1], g0 = gp4[0], g1 = gp4[1];
            aF[8]  = (_Float16)(h0.x + g0.x); aF[9]  = (_Float16)(h0.y + g0.y);
            aF[10] = (_Float16)(h0.z + g0.z); aF[11] = (_Float16)(h0.w + g0.w);
            aF[12] = (_Float16)(h1.x + g1.x); aF[13] = (_Float16)(h1.y + g1.y);
            aF[14] = (_Float16)(h1.z + g1.z); aF[15] = (_Float16)(h1.w + g1.w);
        } else {
#pragma unroll
            for (int j = 8; j < 16; ++j) aF[j] = (_Float16)0.0f;
        }
#pragma unroll
        for (int t = 0; t < NT1; ++t) {
            const v16h bF =
                *(const v16h*)(w1p + (((size_t)kt * NT1 + t) * 32 + lane) * 16);
            acc1[t] = __builtin_amdgcn_wmma_f32_16x16x32_f16(
                          false, aF, false, bF, (short)0, acc1[t], false, false);
        }
    }

    // hidden = relu(acc1 + b1) -> LDS (hoisted row pointers, imm DS offsets)
    float* lrow[8];
#pragma unroll
    for (int i = 0; i < 8; ++i) lrow[i] = &lds_h[wave][mbase + i][col];
#pragma unroll
    for (int t = 0; t < NT1; ++t) {
        const float bias = b1[t * 16 + col];
#pragma unroll
        for (int i = 0; i < 8; ++i) {
            float v = acc1[t][i] + bias;
            v = v > 0.0f ? v : 0.0f;
            lrow[i][t * 16] = v;
        }
    }
    __syncthreads();

    // ---------------- matmul 2: hidden @ w2 ----------------
    v8f acc2[NT2];
#pragma unroll
    for (int t = 0; t < NT2; ++t) acc2[t] = (v8f){};

#pragma unroll
    for (int kt = 0; kt < KT2; ++kt) {
        v16h aF;
        const int kA = kt * 32 + off;
        const int kB = kt * 32 + 16 + off;
        {
            const float4* lp = (const float4*)(&lds_h[wave][arow][kA]);
            float4 l0 = lp[0], l1 = lp[1];
            aF[0] = (_Float16)l0.x; aF[1] = (_Float16)l0.y;
            aF[2] = (_Float16)l0.z; aF[3] = (_Float16)l0.w;
            aF[4] = (_Float16)l1.x; aF[5] = (_Float16)l1.y;
            aF[6] = (_Float16)l1.z; aF[7] = (_Float16)l1.w;
        }
        {
            const float4* lp = (const float4*)(&lds_h[wave][arow][kB]);
            float4 l0 = lp[0], l1 = lp[1];
            aF[8]  = (_Float16)l0.x; aF[9]  = (_Float16)l0.y;
            aF[10] = (_Float16)l0.z; aF[11] = (_Float16)l0.w;
            aF[12] = (_Float16)l1.x; aF[13] = (_Float16)l1.y;
            aF[14] = (_Float16)l1.z; aF[15] = (_Float16)l1.w;
        }
#pragma unroll
        for (int t = 0; t < NT2; ++t) {
            const v16h bF =
                *(const v16h*)(w2p + (((size_t)kt * NT2 + t) * 32 + lane) * 16);
            acc2[t] = __builtin_amdgcn_wmma_f32_16x16x32_f16(
                          false, aF, false, bF, (short)0, acc2[t], false, false);
        }
    }

    // out = acc2 + b2 (+ relu); hoisted row pointers, imm offsets
    float* orow[8];
#pragma unroll
    for (int i = 0; i < 8; ++i)
        orow[i] = hout + (size_t)(r0 + mbase + i) * DOUT + col;
#pragma unroll
    for (int t = 0; t < NT2; ++t) {
        const float bias = b2[t * 16 + col];
#pragma unroll
        for (int i = 0; i < 8; ++i) {
            float v = acc2[t][i] + bias;
            if (RELU_OUT) v = v > 0.0f ? v : 0.0f;
            orow[i][t * 16] = v;
        }
    }
}

// ---------------------------------------------------------------------------
// Small utility kernels (float4-vectorized; all sizes are multiples of 4)
// ---------------------------------------------------------------------------
__global__ void zero_kernel(float4* __restrict__ p, size_t n4) {
    size_t i = (size_t)blockIdx.x * blockDim.x + threadIdx.x;
    if (i < n4) p[i] = (float4){0.0f, 0.0f, 0.0f, 0.0f};
}

__global__ void scale_copy_kernel(const float4* __restrict__ x,
                                  float4* __restrict__ h, float s, size_t n4) {
    size_t i = (size_t)blockIdx.x * blockDim.x + threadIdx.x;
    if (i < n4) {
        float4 v = x[i];
        h[i] = (float4){s * v.x, s * v.y, s * v.z, s * v.w};
    }
}

// ---------------------------------------------------------------------------
// Host launcher
// ---------------------------------------------------------------------------
extern "C" void kernel_launch(void* const* d_in, const int* in_sizes, int n_in,
                              void* d_out, int out_size, void* d_ws, size_t ws_size,
                              hipStream_t stream) {
    (void)n_in; (void)out_size; (void)ws_size;

    const float* x         = (const float*)d_in[0];
    const float* edge_attr = (const float*)d_in[1];
    const int*   esrc      = (const int*)  d_in[2];
    const int*   edst      = (const int*)  d_in[3];
    const float* we[3] = {(const float*)d_in[4],  (const float*)d_in[10], (const float*)d_in[16]};
    const float* be[3] = {(const float*)d_in[5],  (const float*)d_in[11], (const float*)d_in[17]};
    const float* w1[3] = {(const float*)d_in[6],  (const float*)d_in[12], (const float*)d_in[18]};
    const float* b1[3] = {(const float*)d_in[7],  (const float*)d_in[13], (const float*)d_in[19]};
    const float* w2[3] = {(const float*)d_in[8],  (const float*)d_in[14], (const float*)d_in[20]};
    const float* b2[3] = {(const float*)d_in[9],  (const float*)d_in[15], (const float*)d_in[21]};
    const float* rw1 = (const float*)d_in[22];
    const float* rb1 = (const float*)d_in[23];
    const float* rw2 = (const float*)d_in[24];
    const float* rb2 = (const float*)d_in[25];

    const int N = in_sizes[0] / 16;   // 50000
    const int E = in_sizes[2];        // 1600000
    const int Efull = E & ~15;
    const int din[3] = {16, 128, 128};

    // ---- workspace layout (all chunks are multiples of 32B) ----
    float* base = (float*)d_ws;
    auto allocF = [&](size_t nfloat) { float* p = base; base += nfloat; return p; };
    float* hA      = allocF((size_t)N * 128);
    float* hB      = allocF((size_t)N * 128);
    float* agg     = allocF((size_t)N * 128);
    float* phi_pos = allocF((size_t)N * 64);
    float* phi_neg = allocF((size_t)N * 64);

    auto allocH = [&](size_t nhalf) { return (_Float16*)allocF((nhalf + 1) / 2); };
    _Float16 *wep[3], *w1pk[3], *w2pk[3];
    const int dh = 128;
    const int dout_l[3] = {128, 128, 64};
    for (int l = 0; l < 3; ++l) {
        wep[l]  = allocH((size_t)(din[l] / 16) * 512);
        w1pk[l] = allocH((size_t)((din[l] + 31) / 32) * (dh / 16) * 512);
        w2pk[l] = allocH((size_t)(dh / 32) * (dout_l[l] / 16) * 512);
    }
    _Float16* rw1p = allocH((size_t)(64 / 32) * (dh / 16) * 512);
    _Float16* rw2p = allocH((size_t)(dh / 32) * (16 / 16) * 512);

    // ---- pre-pack all weights into WMMA fragment layout ----
    auto pack = [&](const float* W, const float* bias, _Float16* dst,
                    int K, int NC, int KT, int NT, int bias_row) {
        int total = KT * NT * 512;
        pack_frag_kernel<<<(unsigned)((total + 255) / 256), 256, 0, stream>>>(
            W, bias, dst, K, NC, KT, NT, bias_row);
    };
    for (int l = 0; l < 3; ++l) {
        pack(we[l], be[l], wep[l], 16, din[l], 1, din[l] / 16, 16); // bias @ K=16
        pack(w1[l], nullptr, w1pk[l], din[l], dh, (din[l] + 31) / 32, dh / 16, -1);
        pack(w2[l], nullptr, w2pk[l], dh, dout_l[l], dh / 32, dout_l[l] / 16, -1);
    }
    pack(rw1, nullptr, rw1p, 64, dh, 2, dh / 16, -1);
    pack(rw2, nullptr, rw2p, dh, 16, dh / 32, 1, -1);

    // ---- launch geometry ----
    const int ewaves = Efull / 16;
    const dim3 eBlk(256);
    const dim3 eGrd((unsigned)((ewaves * 32 + 255) / 256));
    const int nwaves = (N + 15) / 16;
    const dim3 mBlk(128);
    const dim3 mGrd((unsigned)((nwaves + 3) / 4));
    const int etail = E - Efull;                 // 0 for this workload

    auto zero = [&](float* p, size_t n) {
        size_t n4 = n / 4;
        zero_kernel<<<(unsigned)((n4 + 255) / 256), 256, 0, stream>>>((float4*)p, n4);
    };
    auto edge_layer = [&](int l, const float* hbuf, int dinl) {
        if (dinl == 16)
            edge_msg_kernel<16><<<eGrd, eBlk, 0, stream>>>(hbuf, edge_attr, esrc,
                                                           edst, wep[l], agg, Efull);
        else
            edge_msg_kernel<128><<<eGrd, eBlk, 0, stream>>>(hbuf, edge_attr, esrc,
                                                            edst, wep[l], agg, Efull);
        if (etail > 0) {
            int total = etail * dinl;
            edge_tail_kernel<<<(unsigned)((total + 255) / 256), 256, 0, stream>>>(
                hbuf, edge_attr, esrc, edst, we[l], be[l], agg, dinl, Efull, E);
        }
    };

    for (int pass = 0; pass < 2; ++pass) {
        const float s = (pass == 0) ? 1.0f : -1.0f;
        float* phi_out = (pass == 0) ? phi_pos : phi_neg;

        { // h0 = s * x  [N,16]
            size_t n4 = (size_t)N * 16 / 4;
            scale_copy_kernel<<<(unsigned)((n4 + 255) / 256), 256, 0, stream>>>(
                (const float4*)x, (float4*)hA, s, n4);
        }

        // ---- layer 0: 16 -> 128 -> 128, relu ----
        zero(agg, (size_t)N * 16);
        edge_layer(0, hA, 16);
        node_mlp_kernel<16, 128, 128, true><<<mGrd, mBlk, 0, stream>>>(
            hA, agg, w1pk[0], b1[0], w2pk[0], b2[0], hB, N);

        // ---- layer 1: 128 -> 128 -> 128, relu ----
        zero(agg, (size_t)N * 128);
        edge_layer(1, hB, 128);
        node_mlp_kernel<128, 128, 128, true><<<mGrd, mBlk, 0, stream>>>(
            hB, agg, w1pk[1], b1[1], w2pk[1], b2[1], hA, N);

        // ---- layer 2: 128 -> 128 -> 64, no relu ----
        zero(agg, (size_t)N * 128);
        edge_layer(2, hA, 128);
        node_mlp_kernel<128, 128, 64, false><<<mGrd, mBlk, 0, stream>>>(
            hA, agg, w1pk[2], b1[2], w2pk[2], b2[2], phi_out, N);
    }

    // readout: out = relu((phi_pos + phi_neg) @ rw1 + rb1) @ rw2 + rb2
    node_mlp_kernel<64, 128, 16, false><<<mGrd, mBlk, 0, stream>>>(
        phi_pos, phi_neg, rw1p, rb1, rw2p, rb2, (float*)d_out, N);
}
